// Graph_MFN_15522011807920
// MI455X (gfx1250) — compile-verified, matrix operand order checked by reading, not run on hardware
//
#include <hip/hip_runtime.h>

typedef __bf16 bf16_t;
typedef __attribute__((ext_vector_type(16))) __bf16 v16bf;
typedef __attribute__((ext_vector_type(8)))  float   v8f;
typedef __attribute__((ext_vector_type(4)))  unsigned int u32x4;

#define ACT_NONE 0
#define ACT_RELU 1
#define ACT_SIG  2
#define ACT_TANH 3

// ---------------- model dims ----------------
#define TSEQ 128

// ---------------- bf16 weight region offsets (elements) ----------------
constexpr size_t O_WL    = 0;                       // Wcat_l  [512, 896]  (Wih_l | Whh_l)
constexpr size_t O_WA    = O_WL    + 512*896;       // Wcat_a  [128, 128]  (74 | 32 | pad)
constexpr size_t O_WV    = O_WA    + 128*128;       // Wcat_v  [256, 128]  (35 | 64 | pad)
constexpr size_t O_WTL   = O_WV    + 256*128;       // Wt_l    [128, 256]
constexpr size_t O_WTA   = O_WTL   + 128*256;       // Wt_a    [ 32,  64]
constexpr size_t O_WTV   = O_WTA   + 32*64;         // Wt_v    [ 64, 128]
constexpr size_t O_EW1   = O_WTV   + 64*128;        // e_W1    [112, 224]  (rows 100..111 zero)
constexpr size_t O_EW2   = O_EW1   + 112*224;       // e_W2    [128, 128]  (cols 100..127 zero)
constexpr size_t O_EW3   = O_EW2   + 128*128;       // e_W3    [ 32, 128]  (rows 19..31 zero)
constexpr size_t O_N01W1 = O_EW3   + 32*128;        // [112,160]
constexpr size_t O_N01W2 = O_N01W1 + 112*160;       // [128,128]
constexpr size_t O_N02W1 = O_N01W2 + 128*128;       // [112,192]
constexpr size_t O_N02W2 = O_N02W1 + 112*192;
constexpr size_t O_N12W1 = O_N02W2 + 128*128;       // [112, 96]
constexpr size_t O_N12W2 = O_N12W1 + 112*96;
constexpr size_t O_N012W1= O_N12W2 + 128*128;       // [112,608]
constexpr size_t O_N012W2= O_N012W1+ 112*608;
constexpr size_t O_TNW1  = O_N012W2+ 128*128;       // [112,736]
constexpr size_t O_TNW2  = O_TNW1  + 112*736;
constexpr size_t O_AT1   = O_TNW2  + 128*128;       // att2_W1 [256,128]
constexpr size_t O_AT2   = O_AT1   + 256*128;       // att2_W2 [128,256]
constexpr size_t O_G1W1  = O_AT2   + 128*256;       // [128,256]
constexpr size_t O_G1W2  = O_G1W1  + 128*256;       // [128,128]
constexpr size_t O_G2W1  = O_G1W2  + 128*128;
constexpr size_t O_G2W2  = O_G2W1  + 128*256;
constexpr size_t O_OW1   = O_G2W2  + 128*128;       // out_W1  [128,352]
constexpr size_t BF_TOT  = O_OW1   + 128*352;

// ---------------- f32 region offsets (elements) ----------------
constexpr size_t F_BGL  = 0,            F_BGA  = F_BGL + 512,  F_BGV = F_BGA + 128;
constexpr size_t F_BTL  = F_BGV + 256,  F_BTA  = F_BTL + 128,  F_BTV = F_BTA + 32;
constexpr size_t F_EB1  = F_BTV + 64,   F_EB2  = F_EB1 + 112,  F_EB3 = F_EB2 + 128;
constexpr size_t F_N01B1= F_EB3 + 32,   F_N01B2= F_N01B1 + 112;
constexpr size_t F_N02B1= F_N01B2+128,  F_N02B2= F_N02B1 + 112;
constexpr size_t F_N12B1= F_N02B2+128,  F_N12B2= F_N12B1 + 112;
constexpr size_t F_N012B1=F_N12B2+128,  F_N012B2=F_N012B1 + 112;
constexpr size_t F_TNB1 = F_N012B2+128, F_TNB2 = F_TNB1 + 112;
constexpr size_t F_AT1B = F_TNB2 + 128, F_AT2B = F_AT1B + 256;
constexpr size_t F_G1B1 = F_AT2B + 128, F_G1B2 = F_G1B1 + 128;
constexpr size_t F_G2B1 = F_G1B2 + 128, F_G2B2 = F_G2B1 + 128;
constexpr size_t F_OB1  = F_G2B2 + 128, F_OW2  = F_OB1 + 128,  F_OB2 = F_OW2 + 128;
constexpr size_t F_LAST = F_OB2 + 8;                 // last_hs [256, 352]
constexpr size_t F_TOT  = F_LAST + 256*352;

constexpr size_t SMEM_BYTES = 214016;                // dynamic LDS per WG (< 320KB)

// =====================================================================
// device helpers
// =====================================================================
__device__ inline float sigmf(float x) { return 1.0f / (1.0f + __expf(-x)); }

union FragCast { u32x4 q[2]; v16bf v; };

// explicit global-addrspace pointer so weight frags lower to global_load_b128
typedef const __attribute__((address_space(1))) u32x4* g4p;
// explicit LDS-addrspace pointers so A-frag address math stays in 32-bit LDS
// space (no per-load generic->local addrspacecast null-check cndmask chains)
typedef __attribute__((address_space(3))) const bf16_t* lbf;
typedef __attribute__((address_space(3))) const u32x4* l4p;

// 16-bit fragment (ISA 7.12.2): lane r = M (A) / N (B); half-wave selects K base 0/8;
// vector elems 0..7 = K kb..kb+7, elems 8..15 = K kb+16..kb+23

// LDS fragment (A): base pointer already in addrspace(3)
__device__ inline v16bf load_frag_l(lbf M, int ld, int r0, int k0) {
  int lane = (int)(threadIdx.x & 31);
  lbf p = M + (size_t)(r0 + (lane & 15)) * ld + (size_t)(k0 + ((lane >> 4) << 3));
  FragCast f;
  f.q[0] = *(l4p)(p);
  f.q[1] = *(l4p)(p + 16);
  return f.v;
}

// global (weight) fragment: force addrspace(1) loads
__device__ inline v16bf load_frag_g(const bf16_t* M, int ld, int r0, int k0) {
  int lane = (int)(threadIdx.x & 31);
  const bf16_t* p = M + (size_t)(r0 + (lane & 15)) * ld + (size_t)(k0 + ((lane >> 4) << 3));
  FragCast f;
  f.q[0] = *(g4p)(p);
  f.q[1] = *(g4p)(p + 16);
  return f.v;
}

// C/D 16x16 f32: lane = N (0..15 / 16..31), VGPR j = row (lanes>=16 -> rows 8..15)
__device__ inline void store_tile(float* C, int ldc, int n0, v8f acc, int act) {
  int lane = (int)(threadIdx.x & 31);
  int n  = n0 + (lane & 15);
  int rb = (lane >> 4) << 3;
#pragma unroll
  for (int j = 0; j < 8; ++j) {
    float v = acc[j];
    if (act == ACT_RELU)      v = fmaxf(v, 0.0f);
    else if (act == ACT_SIG)  v = sigmf(v);
    else if (act == ACT_TANH) v = tanhf(v);
    C[(size_t)(rb + j) * ldc + n] = v;
  }
}

// NT register-blocked tile group, compile-time K -> fully unrolled straight-line
// SSA pipeline: fragments for k+32 are issued before the NT WMMAs of the
// current k, copies become register renames, backend emits partial waits.
template <int K, int NT>
__device__ inline void gemm_tiles(const bf16_t* A, int lda,
                                  const bf16_t* W, const float* bias,
                                  float* C, int ldc, int act,
                                  int n0, int stride) {
  const int lane = (int)(threadIdx.x & 31);
  lbf Al = (lbf)A;                            // one generic->LDS cast, hoisted
  v8f acc[NT];
#pragma unroll
  for (int i = 0; i < NT; ++i) {
    float bv = bias ? bias[n0 + i * stride + (lane & 15)] : 0.0f;
    acc[i] = (v8f){ bv, bv, bv, bv, bv, bv, bv, bv };
  }
  v16bf a = load_frag_l(Al, lda, 0, 0);
  v16bf b[NT];
#pragma unroll
  for (int i = 0; i < NT; ++i) b[i] = load_frag_g(W, K, n0 + i * stride, 0);

#pragma unroll
  for (int k0 = 0; k0 < K; k0 += 32) {
    v16bf a_c = a;
    v16bf b_c[NT];
#pragma unroll
    for (int i = 0; i < NT; ++i) b_c[i] = b[i];
    if (k0 + 32 < K) {                        // compile-time after unroll
      a = load_frag_l(Al, lda, 0, k0 + 32);
#pragma unroll
      for (int i = 0; i < NT; ++i) b[i] = load_frag_g(W, K, n0 + i * stride, k0 + 32);
    }
#pragma unroll
    for (int i = 0; i < NT; ++i) {
      acc[i] = __builtin_amdgcn_wmma_f32_16x16x32_bf16(false, a_c, false, b_c[i], (short)0,
                                                       acc[i], false, false);
    }
  }
#pragma unroll
  for (int i = 0; i < NT; ++i) store_tile(C, ldc, n0 + i * stride, acc[i], act);
}

// block GEMM:  C[16 x N] = act( A[16 x K](bf16,lda) @ W[N x K]^T(bf16, ld=K) + bias )
// waves split the N tiles cyclically; each wave register-blocks its tiles (<=4).
template <int K>
__device__ inline void gemm16(const bf16_t* A, int lda,
                              const bf16_t* W, const float* bias,
                              float* C, int ldc, int N, int act) {
  const int wave = (int)(threadIdx.x >> 5);
  const int nwav = (int)(blockDim.x >> 5);
  const int tiles = N >> 4;
  const int own = (wave < tiles) ? ((tiles - 1 - wave) / nwav + 1) : 0;
  const int n0 = wave * 16;
  const int stride = nwav * 16;
  switch (own) {            // wave-uniform: EXEC all-ones inside each case
    case 4: gemm_tiles<K, 4>(A, lda, W, bias, C, ldc, act, n0, stride); break;
    case 3: gemm_tiles<K, 3>(A, lda, W, bias, C, ldc, act, n0, stride); break;
    case 2: gemm_tiles<K, 2>(A, lda, W, bias, C, ldc, act, n0, stride); break;
    case 1: gemm_tiles<K, 1>(A, lda, W, bias, C, ldc, act, n0, stride); break;
    default: break;
  }
}

// dst[16 x wPad] (bf16) <- src[16 x wReal] (f32), cols wReal..wPad zeroed
__device__ inline void pack_pad(bf16_t* dst, int ldd, const float* src, int lds_,
                                int wReal, int wPad) {
  for (int i = (int)threadIdx.x; i < 16 * wPad; i += (int)blockDim.x) {
    int r = i / wPad, c = i - r * wPad;
    dst[(size_t)r * ldd + c] = (c < wReal) ? (bf16_t)src[(size_t)r * lds_ + c] : (bf16_t)0.0f;
  }
}

struct Seg { const float* src; int ld; int w; int ecol; };

// pack concatenated efficacy-scaled segments:  dst[r][off+c] = src[r][c] * eff[r][ecol]
__device__ inline void pack_segs(bf16_t* dst, int ldd, const Seg* segs, int ns,
                                 const float* eff) {
  int off = 0;
  for (int si = 0; si < ns; ++si) {
    Seg sg = segs[si];
    for (int i = (int)threadIdx.x; i < 16 * sg.w; i += (int)blockDim.x) {
      int r = i / sg.w, c = i - r * sg.w;
      dst[(size_t)r * ldd + off + c] =
          (bf16_t)(sg.src[(size_t)r * sg.ld + c] * eff[(size_t)r * 32 + sg.ecol]);
    }
    off += sg.w;
  }
}

// torch LSTMCell gate nonlinearity; also emits [h_old | h_new] bf16 for the
// singleton transform, and commits h <- h_new, c <- c_new.
__device__ inline void lstm_pw(const float* G, int ld4, float* h, float* c,
                               bf16_t* hh, int dh) {
  for (int i = (int)threadIdx.x; i < 16 * dh; i += (int)blockDim.x) {
    int r = i / dh, j = i - r * dh;
    const float* g = G + (size_t)r * ld4;
    float ig = sigmf(g[j]);
    float fg = sigmf(g[dh + j]);
    float gg = tanhf(g[2 * dh + j]);
    float og = sigmf(g[3 * dh + j]);
    float c2 = fg * c[i] + ig * gg;
    float oh = h[i];
    float nh = og * tanhf(c2);
    c[i] = c2;
    hh[(size_t)r * 2 * dh + j]      = (bf16_t)oh;
    hh[(size_t)r * 2 * dh + dh + j] = (bf16_t)nh;
    h[i] = nh;
  }
}

// net2: Linear(K->100[pad112]) -> Linear(100[pad128]->128), no activations
template <int K>
__device__ inline void run_net2(const bf16_t* in,
                                const bf16_t* W1, const float* b1,
                                const bf16_t* W2, const float* b2,
                                float* hid, bf16_t* hidbf, float* out) {
  gemm16<K>(in, 736, W1, b1, hid, 128, 112, ACT_NONE);
  __syncthreads();
  pack_pad(hidbf, 128, hid, 128, 112, 128);
  __syncthreads();
  gemm16<128>(hidbf, 128, W2, b2, out, 128, 128, ACT_NONE);
  __syncthreads();
}

// =====================================================================
// prep kernels (weight conversion / padding / bias fusion)
// =====================================================================
__global__ void zero_b16_kernel(bf16_t* p, int n) {
  int i = blockIdx.x * blockDim.x + threadIdx.x;
  if (i < n) p[i] = (bf16_t)0.0f;
}
__global__ void zero_f32_kernel(float* p, int n) {
  int i = blockIdx.x * blockDim.x + threadIdx.x;
  if (i < n) p[i] = 0.0f;
}
__global__ void cvt_pad_kernel(bf16_t* dst, const float* src, int N, int K,
                               int ldd, int colOff) {
  int i = blockIdx.x * blockDim.x + threadIdx.x;
  if (i < N * K) {
    int n = i / K, k = i - n * K;
    dst[(size_t)n * ldd + colOff + k] = (bf16_t)src[i];
  }
}
__global__ void copy_f32_kernel(float* dst, const float* src, int n) {
  int i = blockIdx.x * blockDim.x + threadIdx.x;
  if (i < n) dst[i] = src[i];
}
__global__ void add2_f32_kernel(float* dst, const float* a, const float* b, int n) {
  int i = blockIdx.x * blockDim.x + threadIdx.x;
  if (i < n) dst[i] = a[i] + b[i];
}

// =====================================================================
// recurrent kernel: one WG per 16-row batch tile, full T loop in LDS
// =====================================================================
__global__ __launch_bounds__(256) void mfn_recurrent_kernel(
    const float* __restrict__ text, const float* __restrict__ audio,
    const float* __restrict__ video,
    const bf16_t* __restrict__ WB, const float* __restrict__ FB,
    float* __restrict__ lasths) {
  extern __shared__ __align__(16) char smem[];
  bf16_t* xh_l  = (bf16_t*)(smem);            // [16,896] x_l | h_l
  bf16_t* xh_a  = (bf16_t*)(smem + 28672);    // [16,128] x_a | h_a | 0
  bf16_t* xh_v  = (bf16_t*)(smem + 32768);    // [16,128] x_v | h_v | 0
  bf16_t* packK = (bf16_t*)(smem + 36864);    // [16,736] shared bf16 pack buffer
  bf16_t* hh_l  = (bf16_t*)(smem + 60416);    // [16,256]
  bf16_t* hh_a  = (bf16_t*)(smem + 68608);    // [16, 64]
  bf16_t* hh_v  = (bf16_t*)(smem + 70656);    // [16,128]
  float*  h_l   = (float*)(smem + 74752);     // [16,128]
  float*  c_l   = (float*)(smem + 82944);
  float*  h_a   = (float*)(smem + 91136);     // [16, 32]
  float*  c_a   = (float*)(smem + 93184);
  float*  h_v   = (float*)(smem + 95232);     // [16, 64]
  float*  c_v   = (float*)(smem + 99328);
  float*  memv  = (float*)(smem + 103424);    // [16,128]
  float*  G     = (float*)(smem + 111616);    // [16,512] gates / att2 hidden
  float*  S     = (float*)(smem + 144384);    // [16,224] singleton concat
  float*  eff   = (float*)(smem + 158720);    // [16, 32] (19 used)
  float*  hid   = (float*)(smem + 160768);    // [16,128]
  bf16_t* hidbf = (bf16_t*)(smem + 168960);   // [16,128]
  float*  o01   = (float*)(smem + 173056);    // [16,128] later: cHat
  float*  o02   = (float*)(smem + 181248);    // later: g1
  float*  o12   = (float*)(smem + 189440);    // later: g2
  float*  o012  = (float*)(smem + 197632);
  float*  att   = (float*)(smem + 205824);    // attended

  const int tid = (int)threadIdx.x;
  const int b0  = (int)blockIdx.x * 16;

  // ----- init state + static pads -----
  for (int i = tid; i < 16 * 128; i += 256) { h_l[i] = 0.f; c_l[i] = 0.f; memv[i] = 0.f; }
  for (int i = tid; i < 16 * 32;  i += 256) { h_a[i] = 0.f; c_a[i] = 0.f; }
  for (int i = tid; i < 16 * 64;  i += 256) { h_v[i] = 0.f; c_v[i] = 0.f; }
  for (int i = tid; i < 16 * 22;  i += 256) { int r = i / 22; xh_a[r * 128 + 106 + (i - r * 22)] = (bf16_t)0.f; }
  for (int i = tid; i < 16 * 29;  i += 256) { int r = i / 29; xh_v[r * 128 +  99 + (i - r * 29)] = (bf16_t)0.f; }
  __syncthreads();

  for (int t = 0; t < TSEQ; ++t) {
    // ---- pack [x_t | h] per modality ----
    for (int i = tid; i < 16 * 768; i += 256) {
      int r = i / 768, d = i - r * 768;
      xh_l[(size_t)r * 896 + d] = (bf16_t)text[((size_t)(b0 + r) * TSEQ + t) * 768 + d];
    }
    for (int i = tid; i < 16 * 128; i += 256) { int r = i >> 7, j = i & 127; xh_l[(size_t)r * 896 + 768 + j] = (bf16_t)h_l[i]; }
    for (int i = tid; i < 16 * 74; i += 256) {
      int r = i / 74, d = i - r * 74;
      xh_a[(size_t)r * 128 + d] = (bf16_t)audio[((size_t)(b0 + r) * TSEQ + t) * 74 + d];
    }
    for (int i = tid; i < 16 * 32; i += 256) { int r = i >> 5, j = i & 31; xh_a[(size_t)r * 128 + 74 + j] = (bf16_t)h_a[i]; }
    for (int i = tid; i < 16 * 35; i += 256) {
      int r = i / 35, d = i - r * 35;
      xh_v[(size_t)r * 128 + d] = (bf16_t)video[((size_t)(b0 + r) * TSEQ + t) * 35 + d];
    }
    for (int i = tid; i < 16 * 64; i += 256) { int r = i >> 6, j = i & 63; xh_v[(size_t)r * 128 + 35 + j] = (bf16_t)h_v[i]; }

    // ---- prefetch next timestep's inputs (overlaps with this step's compute) ----
    if (t + 1 < TSEQ) {
      for (int i = tid; i < 16 * 12; i += 256) {
        int r = i / 12, sgm = i - r * 12;
        __builtin_prefetch(&text[((size_t)(b0 + r) * TSEQ + (t + 1)) * 768 + sgm * 64], 0, 1);
      }
      if (tid < 16) {
        __builtin_prefetch(&audio[((size_t)(b0 + tid) * TSEQ + (t + 1)) * 74], 0, 1);
        __builtin_prefetch(&video[((size_t)(b0 + tid) * TSEQ + (t + 1)) * 35], 0, 1);
      }
    }
    __syncthreads();

    // ---- 3 LSTM cells: fused [x|h] @ [Wih|Whh]^T + (bih+bhh) ----
    gemm16<896>(xh_l, 896, WB + O_WL, FB + F_BGL, G, 512, 512, ACT_NONE);
    __syncthreads();
    lstm_pw(G, 512, h_l, c_l, hh_l, 128);
    __syncthreads();
    gemm16<128>(xh_a, 128, WB + O_WA, FB + F_BGA, G, 128, 128, ACT_NONE);
    __syncthreads();
    lstm_pw(G, 128, h_a, c_a, hh_a, 32);
    __syncthreads();
    gemm16<128>(xh_v, 128, WB + O_WV, FB + F_BGV, G, 256, 256, ACT_NONE);
    __syncthreads();
    lstm_pw(G, 256, h_v, c_v, hh_v, 64);
    __syncthreads();

    // ---- singleton transforms: s_m = relu([h_old|h_new] @ Wt^T + bt) ----
    gemm16<256>(hh_l, 256, WB + O_WTL, FB + F_BTL, S + 0,   224, 128, ACT_RELU);
    gemm16<64> (hh_a,  64, WB + O_WTA, FB + F_BTA, S + 128, 224,  32, ACT_RELU);
    gemm16<128>(hh_v, 128, WB + O_WTV, FB + F_BTV, S + 160, 224,  64, ACT_RELU);
    __syncthreads();

    // ---- efficacy: 3 pure linears ----
    pack_pad(packK, 736, S, 224, 224, 224);
    __syncthreads();
    gemm16<224>(packK, 736, WB + O_EW1, FB + F_EB1, hid, 128, 112, ACT_NONE);
    __syncthreads();
    pack_pad(hidbf, 128, hid, 128, 112, 128);
    __syncthreads();
    gemm16<128>(hidbf, 128, WB + O_EW2, FB + F_EB2, hid, 128, 128, ACT_NONE);
    __syncthreads();
    pack_pad(hidbf, 128, hid, 128, 128, 128);
    __syncthreads();
    gemm16<128>(hidbf, 128, WB + O_EW3, FB + F_EB3, eff, 32, 32, ACT_NONE);
    __syncthreads();

    // ---- pairwise / higher-order fusion nets ----
    { Seg sg[2] = { {S, 224, 128, 0}, {S + 128, 224, 32, 1} };
      pack_segs(packK, 736, sg, 2, eff); __syncthreads();
      run_net2<160>(packK, WB + O_N01W1, FB + F_N01B1, WB + O_N01W2, FB + F_N01B2, hid, hidbf, o01); }
    { Seg sg[2] = { {S, 224, 128, 2}, {S + 160, 224, 64, 3} };
      pack_segs(packK, 736, sg, 2, eff); __syncthreads();
      run_net2<192>(packK, WB + O_N02W1, FB + F_N02B1, WB + O_N02W2, FB + F_N02B2, hid, hidbf, o02); }
    { Seg sg[2] = { {S + 128, 224, 32, 4}, {S + 160, 224, 64, 5} };
      pack_segs(packK, 736, sg, 2, eff); __syncthreads();
      run_net2<96>(packK, WB + O_N12W1, FB + F_N12B1, WB + O_N12W2, FB + F_N12B2, hid, hidbf, o12); }
    { Seg sg[6] = { {S, 224, 128, 6}, {S + 128, 224, 32, 7}, {S + 160, 224, 64, 8},
                    {o01, 128, 128, 9}, {o02, 128, 128, 10}, {o12, 128, 128, 11} };
      pack_segs(packK, 736, sg, 6, eff); __syncthreads();
      run_net2<608>(packK, WB + O_N012W1, FB + F_N012B1, WB + O_N012W2, FB + F_N012B2, hid, hidbf, o012); }
    { Seg sg[7] = { {S, 224, 128, 12}, {S + 128, 224, 32, 13}, {S + 160, 224, 64, 14},
                    {o01, 128, 128, 15}, {o02, 128, 128, 16}, {o12, 128, 128, 17},
                    {o012, 128, 128, 18} };
      pack_segs(packK, 736, sg, 7, eff); __syncthreads();
      run_net2<736>(packK, WB + O_TNW1, FB + F_TNB1, WB + O_TNW2, FB + F_TNB2, hid, hidbf, att); }

    // ---- both = [attended | mem] (bf16); att2 -> cHat ----
    for (int i = tid; i < 16 * 128; i += 256) { int r = i >> 7, c = i & 127; packK[(size_t)r * 736 + c]       = (bf16_t)att[i]; }
    for (int i = tid; i < 16 * 128; i += 256) { int r = i >> 7, c = i & 127; packK[(size_t)r * 736 + 128 + c] = (bf16_t)memv[i]; }
    __syncthreads();
    gemm16<128>(packK, 736, WB + O_AT1, FB + F_AT1B, G, 256, 256, ACT_RELU);
    __syncthreads();
    pack_pad(packK + 256, 736, G, 256, 256, 256);
    __syncthreads();
    gemm16<256>(packK + 256, 736, WB + O_AT2, FB + F_AT2B, o01 /*cHat*/, 128, 128, ACT_TANH);
    __syncthreads();

    // ---- gamma gates on `both` ----
    gemm16<256>(packK, 736, WB + O_G1W1, FB + F_G1B1, hid, 128, 128, ACT_RELU);
    __syncthreads();
    pack_pad(hidbf, 128, hid, 128, 128, 128);
    __syncthreads();
    gemm16<128>(hidbf, 128, WB + O_G1W2, FB + F_G1B2, o02 /*g1*/, 128, 128, ACT_SIG);
    __syncthreads();
    gemm16<256>(packK, 736, WB + O_G2W1, FB + F_G2B1, hid, 128, 128, ACT_RELU);
    __syncthreads();
    pack_pad(hidbf, 128, hid, 128, 128, 128);
    __syncthreads();
    gemm16<128>(hidbf, 128, WB + O_G2W2, FB + F_G2B2, o12 /*g2*/, 128, 128, ACT_SIG);
    __syncthreads();

    // ---- memory update ----
    for (int i = tid; i < 16 * 128; i += 256) memv[i] = o02[i] * memv[i] + o12[i] * o01[i];
    __syncthreads();
  }

  // ---- emit last_hs = [h_l | h_a | h_v | mem] ----
  for (int i = tid; i < 16 * 352; i += 256) {
    int r = i / 352, c = i - r * 352;
    float v;
    if (c < 128)      v = h_l[r * 128 + c];
    else if (c < 160) v = h_a[r * 32 + (c - 128)];
    else if (c < 224) v = h_v[r * 64 + (c - 160)];
    else              v = memv[r * 128 + (c - 224)];
    lasths[(size_t)(b0 + r) * 352 + c] = v;
  }
}

// =====================================================================
// output head: relu(last_hs @ out_W1^T + b1) @ out_W2^T + b2
// =====================================================================
__global__ __launch_bounds__(256) void mfn_output_kernel(
    const float* __restrict__ lasths, const bf16_t* __restrict__ WB,
    const float* __restrict__ FB, float* __restrict__ out) {
  __shared__ __align__(16) bf16_t A[16 * 352];
  __shared__ float hid[16 * 128];
  const int tid = (int)threadIdx.x;
  const int b0  = (int)blockIdx.x * 16;
  for (int i = tid; i < 16 * 352; i += 256) {
    int r = i / 352, c = i - r * 352;
    A[i] = (bf16_t)lasths[(size_t)(b0 + r) * 352 + c];
  }
  __syncthreads();
  gemm16<352>(A, 352, WB + O_OW1, FB + F_OB1, hid, 128, 128, ACT_RELU);
  __syncthreads();
  if (tid < 16) {
    float acc = FB[F_OB2];
    for (int j = 0; j < 128; ++j) acc += hid[tid * 128 + j] * FB[F_OW2 + j];
    out[b0 + tid] = acc;
  }
}

// =====================================================================
// host launcher
// =====================================================================
extern "C" void kernel_launch(void* const* d_in, const int* in_sizes, int n_in,
                              void* d_out, int out_size, void* d_ws, size_t ws_size,
                              hipStream_t stream) {
  (void)in_sizes; (void)n_in; (void)out_size; (void)ws_size;
  const float* text  = (const float*)d_in[0];
  const float* audio = (const float*)d_in[1];
  const float* video = (const float*)d_in[2];
  const float* P[60];
  for (int i = 0; i < 60; ++i) P[i] = (const float*)d_in[3 + i];
  // P index map (params dict insertion order):
  // 0 Wih_l 1 Whh_l 2 bih_l 3 bhh_l 4 Wt_l 5 bt_l
  // 6..11 audio, 12..17 video,
  // 18..21 n01 (W1,b1,W2,b2), 22..25 n02, 26..29 n12, 30..33 n012, 34..37 tnet,
  // 38..43 e (W1,b1,W2,b2,W3,b3), 44..47 att2, 48..51 g1, 52..55 g2, 56..59 out

  bf16_t* wb = (bf16_t*)d_ws;
  float*  fb = (float*)((char*)d_ws + ((BF_TOT * 2 + 255) & ~(size_t)255));

  zero_b16_kernel<<<(int)((BF_TOT + 255) / 256), 256, 0, stream>>>(wb, (int)BF_TOT);
  zero_f32_kernel<<<(int)((F_TOT + 255) / 256), 256, 0, stream>>>(fb, (int)F_TOT);

  auto cvt = [&](int pi, int N, int K, size_t off, int ldd, int colOff) {
    int tot = N * K;
    cvt_pad_kernel<<<(tot + 255) / 256, 256, 0, stream>>>(wb + off, P[pi], N, K, ldd, colOff);
  };
  auto cpy = [&](int pi, int n, size_t off) {
    copy_f32_kernel<<<(n + 255) / 256, 256, 0, stream>>>(fb + off, P[pi], n);
  };

  // fused LSTM weights [Wih | Whh] with K padding
  cvt(0, 512, 768, O_WL, 896, 0);   cvt(1, 512, 128, O_WL, 896, 768);
  cvt(6, 128, 74,  O_WA, 128, 0);   cvt(7, 128, 32,  O_WA, 128, 74);
  cvt(12, 256, 35, O_WV, 128, 0);   cvt(13, 256, 64, O_WV, 128, 35);
  // singleton transforms
  cvt(4, 128, 256, O_WTL, 256, 0);
  cvt(10, 32, 64,  O_WTA, 64, 0);
  cvt(16, 64, 128, O_WTV, 128, 0);
  // efficacy
  cvt(38, 100, 224, O_EW1, 224, 0);
  cvt(40, 128, 100, O_EW2, 128, 0);
  cvt(42, 19, 128,  O_EW3, 128, 0);
  // subset nets
  cvt(18, 100, 160, O_N01W1, 160, 0);  cvt(20, 128, 100, O_N01W2, 128, 0);
  cvt(22, 100, 192, O_N02W1, 192, 0);  cvt(24, 128, 100, O_N02W2, 128, 0);
  cvt(26, 100, 96,  O_N12W1, 96, 0);   cvt(28, 128, 100, O_N12W2, 128, 0);
  cvt(30, 100, 608, O_N012W1, 608, 0); cvt(32, 128, 100, O_N012W2, 128, 0);
  cvt(34, 100, 736, O_TNW1, 736, 0);   cvt(36, 128, 100, O_TNW2, 128, 0);
  // att2 / gates / out
  cvt(44, 256, 128, O_AT1, 128, 0);    cvt(46, 128, 256, O_AT2, 256, 0);
  cvt(48, 128, 256, O_G1W1, 256, 0);   cvt(50, 128, 128, O_G1W2, 128, 0);
  cvt(52, 128, 256, O_G2W1, 256, 0);   cvt(54, 128, 128, O_G2W2, 128, 0);
  cvt(56, 128, 352, O_OW1, 352, 0);

  // fused / copied biases
  add2_f32_kernel<<<2, 256, 0, stream>>>(fb + F_BGL, P[2], P[3], 512);
  add2_f32_kernel<<<1, 256, 0, stream>>>(fb + F_BGA, P[8], P[9], 128);
  add2_f32_kernel<<<1, 256, 0, stream>>>(fb + F_BGV, P[14], P[15], 256);
  cpy(5, 128, F_BTL);   cpy(11, 32, F_BTA);  cpy(17, 64, F_BTV);
  cpy(39, 100, F_EB1);  cpy(41, 128, F_EB2); cpy(43, 19, F_EB3);
  cpy(19, 100, F_N01B1);  cpy(21, 128, F_N01B2);
  cpy(23, 100, F_N02B1);  cpy(25, 128, F_N02B2);
  cpy(27, 100, F_N12B1);  cpy(29, 128, F_N12B2);
  cpy(31, 100, F_N012B1); cpy(33, 128, F_N012B2);
  cpy(35, 100, F_TNB1);   cpy(37, 128, F_TNB2);
  cpy(45, 256, F_AT1B);   cpy(47, 128, F_AT2B);
  cpy(49, 128, F_G1B1);   cpy(51, 128, F_G1B2);
  cpy(53, 128, F_G2B1);   cpy(55, 128, F_G2B2);
  cpy(57, 128, F_OB1);    cpy(58, 128, F_OW2); cpy(59, 1, F_OB2);

  mfn_recurrent_kernel<<<16, 256, SMEM_BYTES, stream>>>(text, audio, video, wb, fb,
                                                        fb + F_LAST);
  mfn_output_kernel<<<16, 256, 0, stream>>>(fb + F_LAST, wb, fb, (float*)d_out);
}
